// MeshRefinementStage_13589276524723
// MI455X (gfx1250) — matrix operand order repaired and checked by reading, not compiled, not run on hardware
//
#include <hip/hip_runtime.h>
#include <stdint.h>

#define V_N   262144
#define E_N   786432
#define CIMG  256
#define HID   128
#define HWIMG 64
#define LAYERS 3
#define DIN   131

typedef __attribute__((ext_vector_type(16))) __bf16 v16bf;
typedef __attribute__((ext_vector_type(8)))  __bf16 v8bf;
typedef __attribute__((ext_vector_type(4)))  __bf16 v4bf;
typedef __attribute__((ext_vector_type(8)))  float  v8f;

__device__ inline v8f zero_v8f() {
    v8f z;
#pragma unroll
    for (int e = 0; e < 8; ++e) z[e] = 0.0f;
    return z;
}

// Native non-returning fp32 atomic add (STOREcnt-tracked fire-and-forget).
__device__ inline void gatomic_fadd(float* p, float v) {
    asm volatile("global_atomic_add_f32 %0, %1, off" : : "v"(p), "v"(v) : "memory");
}

// Coalesced async fill of LDS from global (bf16 payload, verbatim layout).
// Every thread issues 128-bit ASYNC copies; wait ASYNCcnt==0 before barrier.
__device__ inline void async_fill_lds(__bf16* ldsDst, const __bf16* src, int bytes) {
    const unsigned ldsBase = (unsigned)(uintptr_t)ldsDst;   // low 32 bits = LDS offset
    const char* g = (const char*)src;
    for (int off = threadIdx.x * 16; off < bytes; off += blockDim.x * 16) {
        unsigned l = ldsBase + off;
        const char* ga = g + off;
        asm volatile("global_load_async_to_lds_b128 %0, %1, off"
                     : : "v"(l), "v"(ga) : "memory");
    }
    asm volatile("s_wait_asynccnt 0x0" : : : "memory");
}

// ---------------------------------------------------------------------------
// Kernel 0: one-time weight prep: convert fp32 weights to bf16, transposed
// to [n][k] so GEMM blocks can async-copy them verbatim into LDS.
//   bwT: [128][256]   wT: 6 matrices (layer*2+which) of [128][128]
// ---------------------------------------------------------------------------
__global__ __launch_bounds__(256) void k_prep_weights(
    const float* __restrict__ bw, const float* __restrict__ w0,
    const float* __restrict__ w1, __bf16* __restrict__ bwT, __bf16* __restrict__ wT)
{
    const int i = blockIdx.x * blockDim.x + threadIdx.x;
    if (i < HID * CIMG) {
        const int n = i / CIMG, k = i % CIMG;
        bwT[i] = (__bf16)bw[k * HID + n];
    }
    if (i < 6 * HID * HID) {
        const int m = i / (HID * HID);       // layer*2 + which
        const int r = i % (HID * HID);
        const int n = r >> 7, k = r & 127;
        const int layer = m >> 1, which = m & 1;
        const float* w = which ? w1 : w0;
        wT[i] = (__bf16)w[(size_t)layer * DIN * HID + k * HID + n];
    }
}

// ---------------------------------------------------------------------------
// Kernel 1: bilinear sample (border, align_corners) + bottleneck GEMM + relu
// One wave computes a 16(vert) x 128(chan) tile with WMMA bf16, K=256.
// ---------------------------------------------------------------------------
__global__ __launch_bounds__(256) void k_sample_bottleneck(
    const float* __restrict__ img,     // 256*64*64 CHW
    const float* __restrict__ verts,   // V*3
    const __bf16* __restrict__ bwT,    // [128][256] bf16 transposed
    const float* __restrict__ bb,      // 128
    __bf16* __restrict__ F)            // V*128 bf16
{
    __shared__ __align__(32) __bf16 ldsW[HID * CIMG];  // [n][k], 64 KB

    async_fill_lds(ldsW, bwT, HID * CIMG * 2);
    __syncthreads();

    const int lane = threadIdx.x & 31;
    const int wave = threadIdx.x >> 5;
    const int wpb  = blockDim.x >> 5;
    const int kh   = lane >> 4;
    const int lm   = lane & 15;
    const int numTiles = V_N / 16;

    for (int tile = blockIdx.x * wpb + wave; tile < numTiles; tile += gridDim.x * wpb) {
        const int row = tile * 16 + lm;
        const float xc = verts[row * 3 + 0];
        const float yc = verts[row * 3 + 1];
        const float px = (xc + 1.0f) * 0.5f * (HWIMG - 1);
        const float py = (yc + 1.0f) * 0.5f * (HWIMG - 1);
        const float fx = floorf(px), fy = floorf(py);
        const float wx = px - fx, wy = py - fy;
        int x0 = (int)fx; x0 = x0 < 0 ? 0 : (x0 > HWIMG - 1 ? HWIMG - 1 : x0);
        int x1 = x0 + 1;  x1 = x1 > HWIMG - 1 ? HWIMG - 1 : x1;
        int y0 = (int)fy; y0 = y0 < 0 ? 0 : (y0 > HWIMG - 1 ? HWIMG - 1 : y0);
        int y1 = y0 + 1;  y1 = y1 > HWIMG - 1 ? HWIMG - 1 : y1;
        const int o00 = y0 * HWIMG + x0, o01 = y0 * HWIMG + x1;
        const int o10 = y1 * HWIMG + x0, o11 = y1 * HWIMG + x1;
        const float w00 = (1.f - wx) * (1.f - wy), w01 = wx * (1.f - wy);
        const float w10 = (1.f - wx) * wy,         w11 = wx * wy;

        v8f acc[8];
#pragma unroll
        for (int t = 0; t < 8; ++t) acc[t] = zero_v8f();

        const int koff = kh * 8;
        for (int k = 0; k < CIMG; k += 32) {
            v16bf a;
#pragma unroll
            for (int j = 0; j < 8; ++j) {
                const float* p0 = img + (size_t)(k + koff + j) * (HWIMG * HWIMG);
                const float* p1 = img + (size_t)(k + 16 + koff + j) * (HWIMG * HWIMG);
                float s0 = p0[o00] * w00 + p0[o01] * w01 + p0[o10] * w10 + p0[o11] * w11;
                float s1 = p1[o00] * w00 + p1[o01] * w01 + p1[o10] * w10 + p1[o11] * w11;
                a[j]     = (__bf16)s0;
                a[j + 8] = (__bf16)s1;
            }
            // grouped B preload -> WMMA burst (overlap DS with XDL)
#pragma unroll
            for (int tb = 0; tb < 8; tb += 4) {
                v16bf bv[4];
#pragma unroll
                for (int u = 0; u < 4; ++u) {
                    const int n = (tb + u) * 16 + lm;
                    const v8bf b0 = *(const v8bf*)&ldsW[n * CIMG + k + kh * 8];
                    const v8bf b1 = *(const v8bf*)&ldsW[n * CIMG + k + 16 + kh * 8];
#pragma unroll
                    for (int j = 0; j < 8; ++j) { bv[u][j] = b0[j]; bv[u][j + 8] = b1[j]; }
                }
#pragma unroll
                for (int u = 0; u < 4; ++u) {
                    acc[tb + u] = __builtin_amdgcn_wmma_f32_16x16x32_bf16(
                        false, a, false, bv[u], (short)0, acc[tb + u], false, false);
                }
            }
        }

#pragma unroll
        for (int t = 0; t < 8; ++t) {
            const int n = t * 16 + lm;
            const float bias = bb[n];
#pragma unroll
            for (int j = 0; j < 8; ++j) {
                const int r = tile * 16 + j + kh * 8;
                float v = acc[t][j] + bias;
                v = v > 0.f ? v : 0.f;
                F[(size_t)r * HID + n] = (__bf16)v;
            }
        }
    }
}

// ---------------------------------------------------------------------------
// Kernel 2: one GCN layer's two GEMMs: Y = F@W0f + pos@W0p + b0 ; Z likewise.
// ---------------------------------------------------------------------------
__global__ __launch_bounds__(256) void k_layer_gemm(
    const __bf16* __restrict__ F,      // V*128 bf16
    const float* __restrict__ verts,   // V*3
    const __bf16* __restrict__ w0T,    // [128][128] bf16 transposed
    const __bf16* __restrict__ w1T,    // [128][128] bf16 transposed
    const float* __restrict__ w0,      // 131*128 fp32 (pos rows)
    const float* __restrict__ b0,      // 128
    const float* __restrict__ w1,      // 131*128 fp32 (pos rows)
    const float* __restrict__ b1,      // 128
    float* __restrict__ Y,             // V*128
    float* __restrict__ Z)             // V*128
{
    __shared__ __align__(32) __bf16 ldsW0[HID * HID];  // [n][k], 32 KB
    __shared__ __align__(32) __bf16 ldsW1[HID * HID];  // 32 KB

    async_fill_lds(ldsW0, w0T, HID * HID * 2);
    async_fill_lds(ldsW1, w1T, HID * HID * 2);
    __syncthreads();

    const int lane = threadIdx.x & 31;
    const int wave = threadIdx.x >> 5;
    const int wpb  = blockDim.x >> 5;
    const int kh   = lane >> 4;
    const int lm   = lane & 15;
    const int koff = kh * 8;
    const int numTiles = V_N / 16;

    for (int tile = blockIdx.x * wpb + wave; tile < numTiles; tile += gridDim.x * wpb) {
        const __bf16* fRow = F + (size_t)(tile * 16 + lm) * HID;

        const int nextTile = tile + gridDim.x * wpb;
        if (nextTile < numTiles) {
            const __bf16* nf = F + (size_t)(nextTile * 16 + lm) * HID;
            __builtin_prefetch(nf, 0, 1);        // global_prefetch_b8
            __builtin_prefetch(nf + 64, 0, 1);
        }

        v8f accY[8], accZ[8];
#pragma unroll
        for (int t = 0; t < 8; ++t) { accY[t] = zero_v8f(); accZ[t] = zero_v8f(); }

        for (int k = 0; k < HID; k += 32) {
            const v8bf lo = *(const v8bf*)&fRow[k + koff];
            const v8bf hi = *(const v8bf*)&fRow[k + 16 + koff];
            v16bf a;
#pragma unroll
            for (int j = 0; j < 8; ++j) { a[j] = lo[j]; a[j + 8] = hi[j]; }

#pragma unroll
            for (int tb = 0; tb < 8; tb += 4) {
                v16bf bv0[4], bv1[4];
#pragma unroll
                for (int u = 0; u < 4; ++u) {
                    const int n = (tb + u) * 16 + lm;
                    const v8bf p0 = *(const v8bf*)&ldsW0[n * HID + k + kh * 8];
                    const v8bf p1 = *(const v8bf*)&ldsW0[n * HID + k + 16 + kh * 8];
                    const v8bf q0 = *(const v8bf*)&ldsW1[n * HID + k + kh * 8];
                    const v8bf q1 = *(const v8bf*)&ldsW1[n * HID + k + 16 + kh * 8];
#pragma unroll
                    for (int j = 0; j < 8; ++j) {
                        bv0[u][j] = p0[j]; bv0[u][j + 8] = p1[j];
                        bv1[u][j] = q0[j]; bv1[u][j + 8] = q1[j];
                    }
                }
#pragma unroll
                for (int u = 0; u < 4; ++u) {
                    accY[tb + u] = __builtin_amdgcn_wmma_f32_16x16x32_bf16(
                        false, a, false, bv0[u], (short)0, accY[tb + u], false, false);
                    accZ[tb + u] = __builtin_amdgcn_wmma_f32_16x16x32_bf16(
                        false, a, false, bv1[u], (short)0, accZ[tb + u], false, false);
                }
            }
        }

        // position rows (K = 128..130) + bias, then fp32 store
        float vx[8], vy[8], vz[8];
#pragma unroll
        for (int j = 0; j < 8; ++j) {
            const int r = tile * 16 + j + kh * 8;
            vx[j] = verts[r * 3 + 0];
            vy[j] = verts[r * 3 + 1];
            vz[j] = verts[r * 3 + 2];
        }
#pragma unroll
        for (int t = 0; t < 8; ++t) {
            const int n = t * 16 + lm;
            const float by = b0[n], bz = b1[n];
            const float p0y = w0[(HID + 0) * HID + n];
            const float p1y = w0[(HID + 1) * HID + n];
            const float p2y = w0[(HID + 2) * HID + n];
            const float p0z = w1[(HID + 0) * HID + n];
            const float p1z = w1[(HID + 1) * HID + n];
            const float p2z = w1[(HID + 2) * HID + n];
#pragma unroll
            for (int j = 0; j < 8; ++j) {
                const int r = tile * 16 + j + kh * 8;
                Y[(size_t)r * HID + n] =
                    accY[t][j] + vx[j] * p0y + vy[j] * p1y + vz[j] * p2y + by;
                Z[(size_t)r * HID + n] =
                    accZ[t][j] + vx[j] * p0z + vy[j] * p1z + vz[j] * p2z + bz;
            }
        }
    }
}

// ---------------------------------------------------------------------------
// Kernel 3: edge scatter with native f32 atomics, one wave per edge.
// ---------------------------------------------------------------------------
__global__ __launch_bounds__(256) void k_edge_scatter(
    const int* __restrict__ edges, const float* __restrict__ Z, float* __restrict__ Y)
{
    const int e    = (blockIdx.x * blockDim.x + threadIdx.x) >> 5;
    const int lane = threadIdx.x & 31;
    if (e >= E_N) return;
    const int e0 = edges[e * 2 + 0];
    const int e1 = edges[e * 2 + 1];

    const float4 z1 = *(const float4*)&Z[(size_t)e1 * HID + lane * 4];
    float* y0 = &Y[(size_t)e0 * HID + lane * 4];
    gatomic_fadd(y0 + 0, z1.x); gatomic_fadd(y0 + 1, z1.y);
    gatomic_fadd(y0 + 2, z1.z); gatomic_fadd(y0 + 3, z1.w);

    const float4 z0 = *(const float4*)&Z[(size_t)e0 * HID + lane * 4];
    float* y1 = &Y[(size_t)e1 * HID + lane * 4];
    gatomic_fadd(y1 + 0, z0.x); gatomic_fadd(y1 + 1, z0.y);
    gatomic_fadd(y1 + 2, z0.z); gatomic_fadd(y1 + 3, z0.w);
}

// ---------------------------------------------------------------------------
// Kernel 4: F = bf16(relu(Y)); optionally also emit fp32 feats to output.
// ---------------------------------------------------------------------------
__global__ __launch_bounds__(256) void k_relu_to_bf16(
    const float* __restrict__ Y, __bf16* __restrict__ F, float* __restrict__ outFeats)
{
    const int i = blockIdx.x * blockDim.x + threadIdx.x;  // over V*HID/4
    if (i >= V_N * HID / 4) return;
    float4 y = ((const float4*)Y)[i];
    y.x = y.x > 0.f ? y.x : 0.f;
    y.y = y.y > 0.f ? y.y : 0.f;
    y.z = y.z > 0.f ? y.z : 0.f;
    y.w = y.w > 0.f ? y.w : 0.f;
    v4bf f; f[0] = (__bf16)y.x; f[1] = (__bf16)y.y; f[2] = (__bf16)y.z; f[3] = (__bf16)y.w;
    ((v4bf*)F)[i] = f;
    if (outFeats) ((float4*)outFeats)[i] = y;
}

// ---------------------------------------------------------------------------
// Kernel 5: deform = tanh([feats, verts] @ offset_w + offset_b); verts += deform
// ---------------------------------------------------------------------------
__global__ __launch_bounds__(256) void k_offset_head(
    const float* __restrict__ feats, const float* __restrict__ verts,
    const float* __restrict__ ow, const float* __restrict__ ob,
    float* __restrict__ newVerts)
{
    const int v = blockIdx.x * blockDim.x + threadIdx.x;
    if (v >= V_N) return;
    const float* f = feats + (size_t)v * HID;
    float d0 = ob[0], d1 = ob[1], d2 = ob[2];
    for (int k = 0; k < HID; ++k) {
        const float fv = f[k];
        d0 += fv * ow[k * 3 + 0];
        d1 += fv * ow[k * 3 + 1];
        d2 += fv * ow[k * 3 + 2];
    }
    const float vx = verts[v * 3 + 0], vy = verts[v * 3 + 1], vz = verts[v * 3 + 2];
    d0 += vx * ow[128 * 3 + 0] + vy * ow[129 * 3 + 0] + vz * ow[130 * 3 + 0];
    d1 += vx * ow[128 * 3 + 1] + vy * ow[129 * 3 + 1] + vz * ow[130 * 3 + 1];
    d2 += vx * ow[128 * 3 + 2] + vy * ow[129 * 3 + 2] + vz * ow[130 * 3 + 2];
    newVerts[v * 3 + 0] = vx + tanhf(d0);
    newVerts[v * 3 + 1] = vy + tanhf(d1);
    newVerts[v * 3 + 2] = vz + tanhf(d2);
}

// ---------------------------------------------------------------------------
extern "C" void kernel_launch(void* const* d_in, const int* in_sizes, int n_in,
                              void* d_out, int out_size, void* d_ws, size_t ws_size,
                              hipStream_t stream)
{
    const float* x     = (const float*)d_in[0];
    const float* verts = (const float*)d_in[1];
    const int*   edges = (const int*)d_in[2];
    const float* bw    = (const float*)d_in[3];
    const float* bb    = (const float*)d_in[4];
    const float* w0    = (const float*)d_in[5];
    const float* b0    = (const float*)d_in[6];
    const float* w1    = (const float*)d_in[7];
    const float* b1    = (const float*)d_in[8];
    const float* ow    = (const float*)d_in[9];
    const float* ob    = (const float*)d_in[10];

    float* out      = (float*)d_out;
    float* outVerts = out;
    float* outFeats = out + (size_t)V_N * 3;

    char* ws = (char*)d_ws;
    const size_t offF   = 0;
    const size_t offY   = offF + (size_t)V_N * HID * 2;
    const size_t offZ   = offY + (size_t)V_N * HID * 4;
    const size_t offBWT = offZ + (size_t)V_N * HID * 4;
    const size_t offWT  = offBWT + (size_t)HID * CIMG * 2;

    __bf16* F   = (__bf16*)(ws + offF);
    float*  Y   = (float*)(ws + offY);
    float*  Z   = (float*)(ws + offZ);
    __bf16* bwT = (__bf16*)(ws + offBWT);
    __bf16* wT  = (__bf16*)(ws + offWT);

    k_prep_weights<<<(6 * HID * HID + 255) / 256, 256, 0, stream>>>(bw, w0, w1, bwT, wT);
    k_sample_bottleneck<<<512, 256, 0, stream>>>(x, verts, bwT, bb, F);

    for (int i = 0; i < LAYERS; ++i) {
        const __bf16* w0Ti = wT + (size_t)(i * 2 + 0) * HID * HID;
        const __bf16* w1Ti = wT + (size_t)(i * 2 + 1) * HID * HID;
        const float* w0i = w0 + (size_t)i * DIN * HID;
        const float* b0i = b0 + (size_t)i * HID;
        const float* w1i = w1 + (size_t)i * DIN * HID;
        const float* b1i = b1 + (size_t)i * HID;
        k_layer_gemm<<<512, 256, 0, stream>>>(F, verts, w0Ti, w1Ti, w0i, b0i, w1i, b1i, Y, Z);
        k_edge_scatter<<<E_N / 8, 256, 0, stream>>>(edges, Z, Y);
        const bool last = (i == LAYERS - 1);
        k_relu_to_bf16<<<(V_N * HID / 4 + 255) / 256, 256, 0, stream>>>(
            Y, F, last ? outFeats : (float*)nullptr);
    }

    k_offset_head<<<(V_N + 255) / 256, 256, 0, stream>>>(outFeats, verts, ow, ob, outVerts);
}
